// _AlphaTransformerNet_21260088115467
// MI455X (gfx1250) — compile-verified, matrix operand order checked by reading, not run on hardware
//
#include <hip/hip_runtime.h>
#include <hip/hip_bf16.h>
#include <math.h>

#define B_  32
#define T_  512
#define F_  64
#define D_  512
#define H_  8
#define L_  4
#define DH_ 64
#define FF_ 2048

typedef __attribute__((ext_vector_type(16))) _Float16     v16h;
typedef __attribute__((ext_vector_type(8)))  float        v8f;
typedef __attribute__((ext_vector_type(4)))  unsigned int v4u;
typedef __attribute__((ext_vector_type(8)))  int          v8i;
typedef __attribute__((ext_vector_type(4)))  int          v4i;

__device__ __forceinline__ float gelu_f(float z){
  return 0.5f * z * (1.0f + erff(z * 0.70710678118654752440f));
}

// raw LDS byte offset of a __shared__ object (generic -> AS(3) -> int)
__device__ __forceinline__ unsigned lds_addr_of(const void* p){
  return (unsigned)(unsigned long long)
         (__attribute__((address_space(3))) const void*)p;
}

// ---------------------------------------------------------------------------
// WMMA fragment loaders (16-bit A 16x32 layout, CDNA5 ISA 7.12.2):
//   lane: row = lane&15, half = lane>>4
//   VGPR v in 0..3 : K = 8*half + 2v, 2v+1 ; VGPR 4..7 : +16
// ---------------------------------------------------------------------------
__device__ __forceinline__ v16h ld_frag_lds(const _Float16* base, int ld){
  int lane = threadIdx.x & 31;
  int r  = lane & 15;
  int ko = (lane >> 4) << 3;
  const _Float16* p = base + r * ld + ko;
  v16h f;
#pragma unroll
  for (int v = 0; v < 4; ++v){
    f[2*v]         = p[2*v];
    f[2*v + 1]     = p[2*v + 1];
    f[8 + 2*v]     = p[16 + 2*v];
    f[8 + 2*v + 1] = p[16 + 2*v + 1];
  }
  return f;
}

// generic two-stride LDS fragment: element (r,k) at base[r*rs + k*ks]
__device__ __forceinline__ v16h ld_frag_lds2(const _Float16* base, int rs, int ks){
  int lane = threadIdx.x & 31;
  int r  = lane & 15;
  int ko = (lane >> 4) << 3;
  const _Float16* p = base + r * rs;
  v16h f;
#pragma unroll
  for (int v = 0; v < 4; ++v){
    f[2*v]         = p[(ko + 2*v    ) * ks];
    f[2*v + 1]     = p[(ko + 2*v + 1) * ks];
    f[8 + 2*v]     = p[(16 + ko + 2*v    ) * ks];
    f[8 + 2*v + 1] = p[(16 + ko + 2*v + 1) * ks];
  }
  return f;
}

// fragment from global f16, row stride rs halves, contiguous K
__device__ __forceinline__ v16h ld_frag_gh(const _Float16* base, int rs){
  int lane = threadIdx.x & 31;
  int r  = lane & 15;
  int ko = (lane >> 4) << 3;
  const _Float16* p = base + (size_t)r * rs + ko;
  v16h f;
#pragma unroll
  for (int i = 0; i < 8; ++i){
    f[i]     = p[i];
    f[8 + i] = p[16 + i];
  }
  return f;
}

// ---------------------------------------------------------------------------
// TDM: stage a rows x cols f16 tile (row stride strideHalves in global) into
// LDS with an 8-half pad after each 64-half row (LDS pitch 72 halves).
// D# per CDNA5 ISA ch.8; 2D tensor, groups 2/3 zero.
// clang-23 toolchain: 6-arg builtin (g0, g1, g2, g3, g_extra, cpol).
// ---------------------------------------------------------------------------
__device__ __forceinline__ void tdm_load_tile_f16(
    unsigned ldsOff, const _Float16* gsrc,
    unsigned rows, unsigned cols, unsigned strideHalves){
  unsigned long long ga = (unsigned long long)gsrc;
  v4u g0;
  g0[0] = 1u;                                         // count=1, user mode
  g0[1] = ldsOff;                                     // lds_addr (bytes)
  g0[2] = (unsigned)(ga & 0xFFFFFFFFull);             // global_addr[31:0]
  g0[3] = (unsigned)((ga >> 32) & 0x1FFFFFFull)       // global_addr[56:32]
          | (2u << 30);                               // type = 2 (image)
  // group1: data_size=2B, pad_enable, pad_interval=32 DW (enc 4),
  // pad_amount=4 DW (enc 3)
  unsigned w0 = (1u << 16) | (1u << 20) | (4u << 22) | (3u << 25);
  unsigned long long s0 = strideHalves;               // dim0 stride (elements)
  v8i g1;
  g1[0] = (int)w0;
  g1[1] = (int)((cols & 0xFFFFu) << 16);              // tensor_dim0[15:0]
  g1[2] = (int)(((cols >> 16) & 0xFFFFu)              // tensor_dim0[31:16]
          | ((rows & 0xFFFFu) << 16));                // tensor_dim1[15:0]
  g1[3] = (int)(((rows >> 16) & 0xFFFFu)              // tensor_dim1[31:16]
          | (cols << 16));                            // tile_dim0
  g1[4] = (int)(rows & 0xFFFFu);                      // tile_dim1 (tile_dim2=0)
  g1[5] = (int)(unsigned)(s0 & 0xFFFFFFFFull);        // stride0[31:0]
  g1[6] = (int)((s0 >> 32) & 0xFFFFull);              // stride0[47:32]
  g1[7] = 0;
  v4i z4 = (v4i){0, 0, 0, 0};
  v8i z8 = (v8i){0, 0, 0, 0, 0, 0, 0, 0};
  __builtin_amdgcn_tensor_load_to_lds(g0, g1, z4, z4, z8, 0);
}

// ---------------------------------------------------------------------------
// Conv front-end: s = gelu(conv3(x)), l = gelu(conv7(x)), out = concat(s,l)
// ---------------------------------------------------------------------------
__global__ __launch_bounds__(256) void conv_patch_kernel(
    const float* __restrict__ x, const float* __restrict__ ws,
    const float* __restrict__ wl, float* __restrict__ out){
  int bt = blockIdx.x;
  int b = bt / T_, t = bt % T_;
  __shared__ float xw[7 * F_];
  for (int i = threadIdx.x; i < 7 * F_; i += 256){
    int w = i >> 6, f = i & 63;
    int tt = t + w - 3;
    xw[i] = (tt >= 0 && tt < T_) ? x[((size_t)b * T_ + tt) * F_ + f] : 0.0f;
  }
  __syncthreads();
  int c = threadIdx.x;
  float accl = 0.0f, accs = 0.0f;
  for (int tap = 0; tap < 7; ++tap){
    const float* xr = &xw[tap * F_];
    const float* wr = wl + (size_t)(tap * F_) * 256 + c;
#pragma unroll 8
    for (int f = 0; f < F_; ++f) accl += xr[f] * wr[(size_t)f * 256];
  }
  for (int tap = 0; tap < 3; ++tap){
    const float* xr = &xw[(tap + 2) * F_];
    const float* wr = ws + (size_t)(tap * F_) * 256 + c;
#pragma unroll 8
    for (int f = 0; f < F_; ++f) accs += xr[f] * wr[(size_t)f * 256];
  }
  float* o = out + (size_t)bt * D_;
  o[c]       = gelu_f(accs);
  o[256 + c] = gelu_f(accl);
}

// ---------------------------------------------------------------------------
// LayerNorm over rows of 512; one block (128 threads) per row.
// ---------------------------------------------------------------------------
__global__ __launch_bounds__(128) void ln_kernel(
    const float* __restrict__ in, const float* __restrict__ g,
    const float* __restrict__ bb, float* __restrict__ out){
  int row = blockIdx.x;
  int tid = threadIdx.x;
  const float* r = in + (size_t)row * D_;
  float v0 = r[tid], v1 = r[tid + 128], v2 = r[tid + 256], v3 = r[tid + 384];
  __shared__ float red[4];
  float s = v0 + v1 + v2 + v3;
#pragma unroll
  for (int m = 16; m >= 1; m >>= 1) s += __shfl_xor(s, m, 32);
  if ((tid & 31) == 0) red[tid >> 5] = s;
  __syncthreads();
  float mean = (red[0] + red[1] + red[2] + red[3]) * (1.0f / 512.0f);
  __syncthreads();
  float d0 = v0 - mean, d1 = v1 - mean, d2 = v2 - mean, d3 = v3 - mean;
  float q = d0*d0 + d1*d1 + d2*d2 + d3*d3;
#pragma unroll
  for (int m = 16; m >= 1; m >>= 1) q += __shfl_xor(q, m, 32);
  if ((tid & 31) == 0) red[tid >> 5] = q;
  __syncthreads();
  float var = (red[0] + red[1] + red[2] + red[3]) * (1.0f / 512.0f);
  float inv = rsqrtf(var + 1e-5f);
  float* o = out + (size_t)row * D_;
  o[tid]       = d0 * inv * g[tid]       + bb[tid];
  o[tid + 128] = d1 * inv * g[tid + 128] + bb[tid + 128];
  o[tid + 256] = d2 * inv * g[tid + 256] + bb[tid + 256];
  o[tid + 384] = d3 * inv * g[tid + 384] + bb[tid + 384];
}

// ---------------------------------------------------------------------------
// WMMA GEMM: C[M,N] = epi(A[M,K] @ W[N,K]^T + bias[N])
// epilogue: 0 = none(f32), 1 = gelu(f32), 2 = +resid(f32), 3 = f16 store
// 128 threads (4 waves), 64x64 tile, K stepped by 32 through f16 LDS.
// ---------------------------------------------------------------------------
#define LDT 40
__global__ __launch_bounds__(128) void gemm_kernel(
    const float* __restrict__ A, const float* __restrict__ W,
    const float* __restrict__ bias, const float* __restrict__ resid,
    float* __restrict__ C, _Float16* __restrict__ C16,
    int M, int N, int K, int epilogue){
  __shared__ _Float16 As[64 * LDT];
  __shared__ _Float16 Ws[64 * LDT];
  int tid  = threadIdx.x;
  int lane = tid & 31;
  int wave = tid >> 5;
  int m0 = blockIdx.x * 64;
  int n0 = blockIdx.y * 64;

  v8f acc[4];
#pragma unroll
  for (int nt = 0; nt < 4; ++nt) acc[nt] = (v8f){0,0,0,0,0,0,0,0};

  int lrow = tid >> 1;
  int lcol = (tid & 1) << 4;

  for (int k0 = 0; k0 < K; k0 += 32){
    const float* ap = A + (size_t)(m0 + lrow) * K + k0 + lcol;
    const float* wp = W + (size_t)(n0 + lrow) * K + k0 + lcol;
    if (k0 + 32 < K){
      __builtin_prefetch(ap + 32, 0, 3);
      __builtin_prefetch(wp + 32, 0, 3);
    }
    __syncthreads();
    _Float16* ad = &As[lrow * LDT + lcol];
    _Float16* wd = &Ws[lrow * LDT + lcol];
#pragma unroll
    for (int i = 0; i < 16; ++i){
      ad[i] = (_Float16)ap[i];
      wd[i] = (_Float16)wp[i];
    }
    __syncthreads();
    v16h a = ld_frag_lds(&As[(wave * 16) * LDT], LDT);
#pragma unroll
    for (int nt = 0; nt < 4; ++nt){
      v16h bf = ld_frag_lds(&Ws[(nt * 16) * LDT], LDT);
      acc[nt] = __builtin_amdgcn_wmma_f32_16x16x32_f16(
          false, a, false, bf, (short)0, acc[nt], false, false);
    }
  }

  int mb = m0 + wave * 16 + ((lane >> 4) << 3);
  int nc = lane & 15;
#pragma unroll
  for (int nt = 0; nt < 4; ++nt){
    int n = n0 + nt * 16 + nc;
    float bsn = bias[n];
#pragma unroll
    for (int r = 0; r < 8; ++r){
      size_t idx = (size_t)(mb + r) * N + n;
      float v = acc[nt][r] + bsn;
      if (epilogue == 3){
        C16[idx] = (_Float16)v;
      } else {
        if (epilogue == 1)      v = gelu_f(v);
        else if (epilogue == 2) v += resid[idx];
        C[idx] = v;
      }
    }
  }
}

// ---------------------------------------------------------------------------
// Flash attention with ALiBi. qkv: [B*T, 3D] f16 (q|k|v). out: [B*T, D] f32.
// 128-thread blocks; all 4 waves share (b,h); wave 0 stages K/V chunks into
// LDS via the Tensor Data Mover (pad -> 72-half pitch), waits TENSORcnt,
// then everyone computes from LDS.
// ---------------------------------------------------------------------------
#define LDK 72
__global__ __launch_bounds__(128) void attn_kernel(
    const _Float16* __restrict__ qkv, float* __restrict__ outp){
  const int LDQ = 3 * D_;
  int wslot = threadIdx.x >> 5;
  int lane  = threadIdx.x & 31;
  int gw = blockIdx.x * 4 + wslot;
  int mtile = gw & (T_ / 16 - 1);
  int bh = gw >> 5;            // same for all 4 waves of the block
  int hh = bh & (H_ - 1);
  int b  = bh >> 3;

  __shared__ _Float16 Kbuf[32 * LDK];
  __shared__ _Float16 Vbuf[32 * LDK];
  __shared__ _Float16 Pbuf[4][16 * LDT];

  const _Float16* qbase = qkv + ((size_t)b * T_ + mtile * 16) * LDQ + hh * DH_;
  v16h q0 = ld_frag_gh(qbase,      LDQ);   // d = 0..31
  v16h q1 = ld_frag_gh(qbase + 32, LDQ);   // d = 32..63

  float mrow[8], lrow[8];
  v8f oacc[4];
#pragma unroll
  for (int r = 0; r < 8; ++r){ mrow[r] = -1e30f; lrow[r] = 0.0f; }
#pragma unroll
  for (int nt = 0; nt < 4; ++nt) oacc[nt] = (v8f){0,0,0,0,0,0,0,0};

  const float slope = exp2f(-(float)(hh + 1));   // exact for H=8
  const float scale = 0.125f;
  int hf = lane >> 4;
  int nc = lane & 15;

  unsigned kOff = lds_addr_of(Kbuf);
  unsigned vOff = lds_addr_of(Vbuf);

  for (int kt = 0; kt < T_; kt += 32){
    __syncthreads();                         // prior reads done
    if (wslot == 0){
      const _Float16* kg = qkv + ((size_t)b * T_ + kt) * LDQ + D_     + hh * DH_;
      const _Float16* vg = qkv + ((size_t)b * T_ + kt) * LDQ + 2 * D_ + hh * DH_;
      tdm_load_tile_f16(kOff, kg, 32, 64, LDQ);
      tdm_load_tile_f16(vOff, vg, 32, 64, LDQ);
      __builtin_amdgcn_s_wait_tensorcnt(0);
    }
    __syncthreads();                         // staged data visible

    v8f s0 = (v8f){0,0,0,0,0,0,0,0};
    v8f s1 = (v8f){0,0,0,0,0,0,0,0};
    {
      const _Float16* kb0 = Kbuf;
      const _Float16* kb1 = Kbuf + 16 * LDK;
      s0 = __builtin_amdgcn_wmma_f32_16x16x32_f16(false, q0, false,
              ld_frag_lds(kb0,      LDK), (short)0, s0, false, false);
      s0 = __builtin_amdgcn_wmma_f32_16x16x32_f16(false, q1, false,
              ld_frag_lds(kb0 + 32, LDK), (short)0, s0, false, false);
      s1 = __builtin_amdgcn_wmma_f32_16x16x32_f16(false, q0, false,
              ld_frag_lds(kb1,      LDK), (short)0, s1, false, false);
      s1 = __builtin_amdgcn_wmma_f32_16x16x32_f16(false, q1, false,
              ld_frag_lds(kb1 + 32, LDK), (short)0, s1, false, false);
    }

    float p0[8], p1[8], cf[8];
#pragma unroll
    for (int r = 0; r < 8; ++r){
      int tq = mtile * 16 + hf * 8 + r;
      float t0 = s0[r] * scale - slope * fabsf((float)(tq - (kt + nc)));
      float t1 = s1[r] * scale - slope * fabsf((float)(tq - (kt + 16 + nc)));
      float mx = fmaxf(t0, t1);
#pragma unroll
      for (int m = 8; m >= 1; m >>= 1) mx = fmaxf(mx, __shfl_xor(mx, m, 32));
      float mn = fmaxf(mrow[r], mx);
      cf[r] = expf(mrow[r] - mn);
      float e0 = expf(t0 - mn), e1 = expf(t1 - mn);
      float sm = e0 + e1;
#pragma unroll
      for (int m = 8; m >= 1; m >>= 1) sm += __shfl_xor(sm, m, 32);
      lrow[r] = lrow[r] * cf[r] + sm;
      mrow[r] = mn;
      p0[r] = e0; p1[r] = e1;
    }
#pragma unroll
    for (int nt = 0; nt < 4; ++nt)
#pragma unroll
      for (int r = 0; r < 8; ++r) oacc[nt][r] *= cf[r];

    _Float16* pb = Pbuf[wslot];
#pragma unroll
    for (int r = 0; r < 8; ++r){
      int m = hf * 8 + r;
      pb[m * LDT + nc]      = (_Float16)p0[r];
      pb[m * LDT + 16 + nc] = (_Float16)p1[r];
    }
    __builtin_amdgcn_wave_barrier();         // same-wave LDS exchange
    v16h pf = ld_frag_lds(pb, LDT);
#pragma unroll
    for (int nt = 0; nt < 4; ++nt){
      oacc[nt] = __builtin_amdgcn_wmma_f32_16x16x32_f16(false, pf, false,
                    ld_frag_lds2(Vbuf + nt * 16, 1, LDK), (short)0,
                    oacc[nt], false, false);
    }
  }

#pragma unroll
  for (int nt = 0; nt < 4; ++nt)
#pragma unroll
    for (int r = 0; r < 8; ++r){
      int tq = mtile * 16 + hf * 8 + r;
      outp[((size_t)b * T_ + tq) * D_ + hh * DH_ + nt * 16 + nc] =
          oacc[nt][r] / lrow[r];
    }
}

// ---------------------------------------------------------------------------
// Head: final LN on last-token rows only, D->256 gelu, 256->3, softmax.
// ---------------------------------------------------------------------------
__global__ __launch_bounds__(256) void head_kernel(
    const float* __restrict__ h, const float* __restrict__ fg,
    const float* __restrict__ fb, const float* __restrict__ w1,
    const float* __restrict__ b1, const float* __restrict__ w2,
    const float* __restrict__ b2, float* __restrict__ out){
  int b = blockIdx.x;
  int tid = threadIdx.x;
  __shared__ float hn[D_];
  __shared__ float f1[256];
  __shared__ float red[8];
  __shared__ float logits[3];
  const float* row = h + ((size_t)b * T_ + (T_ - 1)) * D_;
  float a0 = row[tid], a1 = row[tid + 256];
  float s = a0 + a1;
#pragma unroll
  for (int m = 16; m >= 1; m >>= 1) s += __shfl_xor(s, m, 32);
  if ((tid & 31) == 0) red[tid >> 5] = s;
  __syncthreads();
  float mean = 0.0f;
  for (int i = 0; i < 8; ++i) mean += red[i];
  mean *= (1.0f / 512.0f);
  __syncthreads();
  float d0 = a0 - mean, d1 = a1 - mean;
  float q = d0*d0 + d1*d1;
#pragma unroll
  for (int m = 16; m >= 1; m >>= 1) q += __shfl_xor(q, m, 32);
  if ((tid & 31) == 0) red[tid >> 5] = q;
  __syncthreads();
  float var = 0.0f;
  for (int i = 0; i < 8; ++i) var += red[i];
  var *= (1.0f / 512.0f);
  float inv = rsqrtf(var + 1e-5f);
  hn[tid]       = d0 * inv * fg[tid]       + fb[tid];
  hn[tid + 256] = d1 * inv * fg[tid + 256] + fb[tid + 256];
  __syncthreads();
  float acc = b1[tid];
  const float* wr = w1 + (size_t)tid * D_;
#pragma unroll 8
  for (int k = 0; k < D_; ++k) acc += hn[k] * wr[k];
  f1[tid] = gelu_f(acc);
  __syncthreads();
  if (tid < 3){
    float a2 = b2[tid];
    const float* w2r = w2 + (size_t)tid * 256;
    for (int j = 0; j < 256; ++j) a2 += f1[j] * w2r[j];
    logits[tid] = a2;
  }
  __syncthreads();
  if (tid == 0){
    float mx = fmaxf(logits[0], fmaxf(logits[1], logits[2]));
    float e0 = expf(logits[0] - mx), e1 = expf(logits[1] - mx),
          e2 = expf(logits[2] - mx);
    float den = 1.0f / (e0 + e1 + e2);
    out[(size_t)b * 3 + 0] = e0 * den;
    out[(size_t)b * 3 + 1] = e1 * den;
    out[(size_t)b * 3 + 2] = e2 * den;
  }
}

// ---------------------------------------------------------------------------
extern "C" void kernel_launch(void* const* d_in, const int* in_sizes, int n_in,
                              void* d_out, int out_size, void* d_ws, size_t ws_size,
                              hipStream_t stream) {
  (void)in_sizes; (void)n_in; (void)out_size; (void)ws_size;
  const float* x        = (const float*)d_in[0];
  const float* conv_s_w = (const float*)d_in[1];
  const float* conv_l_w = (const float*)d_in[2];
  const float* patch_g  = (const float*)d_in[3];
  const float* patch_b  = (const float*)d_in[4];
  const float* ln1_g    = (const float*)d_in[5];
  const float* ln1_b    = (const float*)d_in[6];
  const float* in_w     = (const float*)d_in[7];
  const float* in_b     = (const float*)d_in[8];
  const float* out_w    = (const float*)d_in[9];
  const float* out_b    = (const float*)d_in[10];
  const float* ln2_g    = (const float*)d_in[11];
  const float* ln2_b    = (const float*)d_in[12];
  const float* ffn_w1   = (const float*)d_in[13];
  const float* ffn_b1   = (const float*)d_in[14];
  const float* ffn_w2   = (const float*)d_in[15];
  const float* ffn_b2   = (const float*)d_in[16];
  const float* final_g  = (const float*)d_in[17];
  const float* final_b  = (const float*)d_in[18];
  const float* head_w1  = (const float*)d_in[19];
  const float* head_b1  = (const float*)d_in[20];
  const float* head_w2  = (const float*)d_in[21];
  const float* head_b2  = (const float*)d_in[22];

  const size_t BT = (size_t)B_ * T_;
  char* base = (char*)d_ws;
  float*     h    = (float*)(base);                         // 32 MB
  float*     xn   = (float*)(base + BT * D_ * 4);           // 32 MB (also attn out)
  _Float16*  qkvh = (_Float16*)(base + 2 * BT * D_ * 4);    // 48 MB f16 (+conv scratch)
  float*     t1   = (float*)(base + 2 * BT * D_ * 4 + BT * 3 * D_ * 2); // 128 MB

  // conv + gelu + concat -> pre-LN activations (scratch in qkvh region, f32)
  conv_patch_kernel<<<B_ * T_, 256, 0, stream>>>(x, conv_s_w, conv_l_w,
                                                 (float*)qkvh);
  ln_kernel<<<B_ * T_, 128, 0, stream>>>((float*)qkvh, patch_g, patch_b, h);

  for (int l = 0; l < L_; ++l){
    ln_kernel<<<B_ * T_, 128, 0, stream>>>(h, ln1_g + (size_t)l * D_,
                                           ln1_b + (size_t)l * D_, xn);
    gemm_kernel<<<dim3(BT / 64, (3 * D_) / 64), 128, 0, stream>>>(
        xn, in_w + (size_t)l * 3 * D_ * D_, in_b + (size_t)l * 3 * D_,
        nullptr, nullptr, qkvh, (int)BT, 3 * D_, D_, 3);
    attn_kernel<<<(B_ * H_ * (T_ / 16)) / 4, 128, 0, stream>>>(qkvh, xn);
    gemm_kernel<<<dim3(BT / 64, D_ / 64), 128, 0, stream>>>(
        xn, out_w + (size_t)l * D_ * D_, out_b + (size_t)l * D_,
        h, h, nullptr, (int)BT, D_, D_, 2);
    ln_kernel<<<B_ * T_, 128, 0, stream>>>(h, ln2_g + (size_t)l * D_,
                                           ln2_b + (size_t)l * D_, xn);
    gemm_kernel<<<dim3(BT / 64, FF_ / 64), 128, 0, stream>>>(
        xn, ffn_w1 + (size_t)l * FF_ * D_, ffn_b1 + (size_t)l * FF_,
        nullptr, t1, nullptr, (int)BT, FF_, D_, 1);
    gemm_kernel<<<dim3(BT / 64, D_ / 64), 128, 0, stream>>>(
        t1, ffn_w2 + (size_t)l * D_ * FF_, ffn_b2 + (size_t)l * D_,
        h, h, nullptr, (int)BT, D_, FF_, 2);
  }

  head_kernel<<<B_, 256, 0, stream>>>(h, final_g, final_b, head_w1, head_b1,
                                      head_w2, head_b2, (float*)d_out);
}
